// LSNN_NHDN_35313221107907
// MI455X (gfx1250) — matrix operand order repaired and checked by reading, not compiled, not run on hardware
//
#include <hip/hip_runtime.h>

typedef __attribute__((ext_vector_type(16))) _Float16 v16h;
typedef __attribute__((ext_vector_type(8)))  float    v8f;

#define B_   512
#define T_   2048
#define D_   128
#define C_   10
#define VTHR 1.0f
#define VDEC 0.9f
#define MDEC 0.9f

// T-chunking for the LIF scan (phase 1)
#define CH1   512          // steps per chunk
#define NCH1  (T_ / CH1)   // 4 chunks
#define WARM  256          // warm-up steps; 0.9^256 ~ 2e-12 -> bit-convergent

// T-chunking for the leaky-integrator scan (phase 3)
#define CH3   64
#define NCH3  (T_ / CH3)   // 32 chunks
#define DEC64 0.0011790184577738583f   // 0.9^64
#define LOG2_MDEC -0.15200309344504995f

union AFrag { unsigned int u[8]; v16h h; };

static __device__ __forceinline__ unsigned int f16bits(float f) {
    _Float16 hv = (_Float16)f;
    union { _Float16 h; unsigned short u; } cv; cv.h = hv;
    return (unsigned int)cv.u;
}

// ---------------------------------------------------------------------------
// Phase 1: LIF encoder scan, one thread per (chunk, batch, dim), on+off
// channels per thread. 256-step warm-up from v=0 makes chunked scans
// bit-converge to the exact serial scan (decay 0.9 + reset-to-zero).
// Spikes bit-packed with __ballot: row (b,t) = 8 dwords (on: 0..3, off: 4..7).
// ---------------------------------------------------------------------------
__global__ void lif_encode_kernel(const float* __restrict__ x,
                                  unsigned int* __restrict__ spikes) {
    const int g    = blockIdx.x * blockDim.x + threadIdx.x;  // NCH1*B_*D_ threads
    const int c    = g >> 16;            // / (B_*D_)
    const int rem  = g & 0xFFFF;
    const int b    = rem >> 7;           // / D_
    const int d    = rem & (D_ - 1);
    const int lane = threadIdx.x & 31;
    const int w    = d >> 5;             // channel word 0..3

    const float* xp = x + (size_t)b * T_ * D_ + d;
    unsigned int* row = spikes + (size_t)b * T_ * 8;

    const int t0   = c * CH1;
    const int warm = (c == 0) ? 0 : WARM;

    float von = 0.f, voff = 0.f;

    // Warm-up: run the LIF dynamics, emit nothing.
    for (int t = t0 - warm; t < t0; t += 4) {
        float xv[4];
#pragma unroll
        for (int k = 0; k < 4; ++k) xv[k] = xp[(size_t)(t + k) * D_];
#pragma unroll
        for (int k = 0; k < 4; ++k) {
            von  = VDEC * von  + xv[k];
            voff = VDEC * voff - xv[k];
            von  = (von  > VTHR) ? 0.f : von;
            voff = (voff > VTHR) ? 0.f : voff;
        }
    }

    // Main chunk: emit ballot-packed spikes.
    for (int t = t0; t < t0 + CH1; t += 8) {
        float xv[8];
#pragma unroll
        for (int k = 0; k < 8; ++k) xv[k] = xp[(size_t)(t + k) * D_];
        if (t + 32 < T_)
            __builtin_prefetch(xp + (size_t)(t + 32) * D_, 0, 1);
#pragma unroll
        for (int k = 0; k < 8; ++k) {
            von  = VDEC * von  + xv[k];
            voff = VDEC * voff - xv[k];
            const int son  = von  > VTHR;
            const int soff = voff > VTHR;
            von  = son  ? 0.f : von;
            voff = soff ? 0.f : voff;
            const unsigned int mon  = (unsigned int)__ballot(son);
            const unsigned int moff = (unsigned int)__ballot(soff);
            if (lane == 0) {
                row[(size_t)(t + k) * 8 + w]     = mon;
                row[(size_t)(t + k) * 8 + 4 + w] = moff;
            }
        }
    }
}

// ---------------------------------------------------------------------------
// Phase 2: y[b,t,:] = spikes[b,t,:] @ W + bias via v_wmma_f32_16x16x32_f16.
// One wave per tile = (batch row b, 16 consecutive timesteps). M = timestep,
// N = class, K = 256 channels (8 fragments). Spike bits expanded in-register
// into the 16-bit A-matrix 16x32 VGPR layout.
// ---------------------------------------------------------------------------
__global__ void spike_matmul_kernel(const unsigned int* __restrict__ spikes,
                                    const float* __restrict__ W,
                                    const float* __restrict__ bias,
                                    float* __restrict__ out) {
    const int lane   = threadIdx.x & 31;
    const int waveId = blockIdx.x * (blockDim.x >> 5) + (threadIdx.x >> 5);
    const int nWaves = gridDim.x * (blockDim.x >> 5);
    const int r = lane & 15;   // A row within tile / output column / W column
    const int h = lane >> 4;   // lane-half selects K sub-range

    // W (B-matrix) f16 fragments, built once per wave (L1-cached loads).
    // B 32x16 layout: lane half h holds K=8h..8h+7 and 16+8h..16+8h+7,
    // dword i packs rows K = 8h+2i(+8 if i>=4), +1 for column n = lane%16.
    AFrag wf[8];
#pragma unroll
    for (int f = 0; f < 8; ++f) {
#pragma unroll
        for (int i = 0; i < 8; ++i) {
            const int k = 32 * f + 8 * h + 2 * i + (i >= 4 ? 8 : 0);
            const unsigned int lo = (r < C_) ? f16bits(W[(size_t)k * C_ + r])       : 0u;
            const unsigned int hi = (r < C_) ? f16bits(W[(size_t)(k + 1) * C_ + r]) : 0u;
            wf[f].u[i] = lo | (hi << 16);
        }
    }
    const float bval = (r < C_) ? bias[r] : 0.f;

    const int nTiles = B_ * (T_ / 16);
    for (int tile = waveId; tile < nTiles; tile += nWaves) {
        const int b  = tile >> 7;          // T_/16 == 128 tiles per batch row
        const int t0 = (tile & 127) << 4;

        // Each lane loads the 32-byte bit-packed spike row for timestep t0+r.
        const unsigned int* sp = spikes + ((size_t)b * T_ + t0 + r) * 8;
        const uint4 lo4 = *(const uint4*)(sp);
        const uint4 hi4 = *(const uint4*)(sp + 4);
        const unsigned int wrd[8] = {lo4.x, lo4.y, lo4.z, lo4.w,
                                     hi4.x, hi4.y, hi4.z, hi4.w};

        v8f cacc = {};
#pragma unroll
        for (int f = 0; f < 8; ++f) {
            // lane half h needs K = 32f+8h+{0..7} and 32f+16+8h+{0..7}
            const unsigned int lob = (wrd[f] >> (8 * h)) & 0xFFu;
            const unsigned int hib = (wrd[f] >> (16 + 8 * h)) & 0xFFu;
            AFrag a;
#pragma unroll
            for (int i = 0; i < 4; ++i) {
                a.u[i]     = ((lob >> (2 * i)) & 1u) * 0x3C00u |
                             ((lob >> (2 * i + 1)) & 1u) * 0x3C000000u;
                a.u[4 + i] = ((hib >> (2 * i)) & 1u) * 0x3C00u |
                             ((hib >> (2 * i + 1)) & 1u) * 0x3C000000u;
            }
            cacc = __builtin_amdgcn_wmma_f32_16x16x32_f16(
                       false, a.h, false, wf[f].h, (short)0, cacc, false, false);
        }

        // C/D layout: VGPR j holds M = j + 8h, N = lane%16.
        if (r < C_) {
#pragma unroll
            for (int j = 0; j < 8; ++j) {
                out[((size_t)b * T_ + (t0 + j + 8 * h)) * C_ + r] = cacc[j] + bval;
            }
        }
    }
}

// ---------------------------------------------------------------------------
// Phase 3A: chunk-local leaky scan over CH3=64 steps, entirely in registers
// (64 loads issued up-front for MLP). Writes local values in place and the
// chunk-final local sum into the carry scratch.
// ---------------------------------------------------------------------------
__global__ void scanA_kernel(float* __restrict__ out,
                             float* __restrict__ carries) {
    const int tid = blockIdx.x * blockDim.x + threadIdx.x;   // B_*C_*NCH3
    if (tid >= B_ * C_ * NCH3) return;
    const int b     = tid / (C_ * NCH3);
    const int rr    = tid % (C_ * NCH3);
    const int chunk = rr / C_;
    const int cc    = rr % C_;

    float* p = out + ((size_t)b * T_ + (size_t)chunk * CH3) * C_ + cc;
    float y[CH3];
#pragma unroll
    for (int k = 0; k < CH3; ++k) y[k] = p[(size_t)k * C_];
    float m = 0.f;
#pragma unroll
    for (int k = 0; k < CH3; ++k) { m = MDEC * m + y[k]; y[k] = m; }
#pragma unroll
    for (int k = 0; k < CH3; ++k) p[(size_t)k * C_] = y[k];

    carries[((size_t)b * C_ + cc) * NCH3 + chunk] = m;   // local end value
}

// ---------------------------------------------------------------------------
// Phase 3B: per (b,c), turn chunk-end local sums into exclusive carry-ins:
// carryIn_0 = 0; carryIn_j = L_{j-1} + 0.9^64 * carryIn_{j-1}.  (in place)
// ---------------------------------------------------------------------------
__global__ void scanB_kernel(float* __restrict__ carries) {
    const int tid = blockIdx.x * blockDim.x + threadIdx.x;   // B_*C_
    if (tid >= B_ * C_) return;
    float* p = carries + (size_t)tid * NCH3;
    float cin = 0.f;
    for (int j = 0; j < NCH3; ++j) {
        const float L = p[j];
        p[j] = cin;
        cin = L + DEC64 * cin;
    }
}

// ---------------------------------------------------------------------------
// Phase 3C: elementwise fix-up: out[b, t, c] += carryIn * 0.9^{(t%64)+1}.
// ---------------------------------------------------------------------------
__global__ void scanC_kernel(float* __restrict__ out,
                             const float* __restrict__ carries) {
    const size_t e = (size_t)blockIdx.x * blockDim.x + threadIdx.x;
    if (e >= (size_t)B_ * T_ * C_) return;
    const int b  = (int)(e / ((size_t)T_ * C_));
    const int rr = (int)(e % ((size_t)T_ * C_));
    const int t  = rr / C_;
    const int cc = rr % C_;
    const int chunk = t >> 6;
    const int tl    = t & (CH3 - 1);
    const float cin = carries[((size_t)b * C_ + cc) * NCH3 + chunk];
    out[e] += cin * exp2f((float)(tl + 1) * LOG2_MDEC);
}

extern "C" void kernel_launch(void* const* d_in, const int* in_sizes, int n_in,
                              void* d_out, int out_size, void* d_ws, size_t ws_size,
                              hipStream_t stream) {
    const float* x    = (const float*)d_in[0];   // (B, T, D) f32
    const float* W    = (const float*)d_in[1];   // (2D, C) f32
    const float* bias = (const float*)d_in[2];   // (C,) f32
    float* out = (float*)d_out;                  // (B, T, C) f32

    unsigned int* spikes = (unsigned int*)d_ws;               // 32 MiB
    float* carries = (float*)((char*)d_ws + (size_t)B_ * T_ * 8 * 4);  // 640 KiB

    // Phase 1: chunked LIF scan with warm-up. 262144 threads (8192 waves).
    lif_encode_kernel<<<(NCH1 * B_ * D_) / 256, 256, 0, stream>>>(x, spikes);

    // Phase 2: WMMA spike matmul. 8192 waves x 8 tiles (grid-stride).
    spike_matmul_kernel<<<1024, 256, 0, stream>>>(spikes, W, bias, out);

    // Phase 3: parallel linear scan (local / carry / fix-up).
    scanA_kernel<<<(B_ * C_ * NCH3 + 255) / 256, 256, 0, stream>>>(out, carries);
    scanB_kernel<<<(B_ * C_ + 255) / 256, 256, 0, stream>>>(carries);
    {
        const size_t nElem = (size_t)B_ * T_ * C_;
        scanC_kernel<<<(unsigned)((nElem + 255) / 256), 256, 0, stream>>>(out, carries);
    }
}
